// SAM_MetaC_With_Depth_Attention_2way_dpt_inf_pe_62603443306763
// MI455X (gfx1250) — compile-verified, matrix operand order checked by reading, not run on hardware
//
#include <hip/hip_runtime.h>
#include <math.h>

// ---------------------------------------------------------------------------
// SAM-style cross-attention block (rgb<-dpt) + depth positional embedding,
// MI455X (gfx1250, wave32).
//
// All big matmuls run on v_wmma_f32_16x16x32_f16 (f32 accumulate). Tiles are
// stored in LDS in *fragment-major* order ([row][K], rows padded to 40 halves)
// so every WMMA fragment is fetched as two 16-byte ds_load_b128 instead of
// 16 scalar ds_load_u16. Attention is flash-style (no 8x4096x4096 scores).
// ---------------------------------------------------------------------------

typedef __attribute__((ext_vector_type(16))) _Float16 v16h;
typedef __attribute__((ext_vector_type(8)))  _Float16 v8h;
typedef __attribute__((ext_vector_type(8)))  float    v8f;
typedef __attribute__((ext_vector_type(4)))  float    v4f;

#define DIMC 256
#define NPIX 4096
#define HEADS 8
#define HD 32
#define MLPC 1024
#define LN_EPS 1e-5f
#define ROWH 40  // padded LDS row length (halves) for 32-K fragment rows

// K-index mapping for 16-bit A/B fragments (ISA 7.12.2, 16-bit 16x32 layout):
// lanes 0-15: halves 0..7 -> K0..7,  halves 8..15 -> K16..23
// lanes 16-31: halves 0..7 -> K8..15, halves 8..15 -> K24..31
// => a lane's fragment is two contiguous 8-half runs at K=hi and K=hi+16.
__device__ __forceinline__ v16h load_frag_row(const _Float16* __restrict__ row,
                                              int lane) {
  int hi = (lane & 16) ? 8 : 0;
  v8h lo = *(const v8h*)(row + hi);
  v8h hs = *(const v8h*)(row + hi + 16);
  return __builtin_shufflevector(lo, hs, 0, 1, 2, 3, 4, 5, 6, 7,
                                 8, 9, 10, 11, 12, 13, 14, 15);
}

__device__ __forceinline__ v8f wmma_f16(const v16h& a, const v16h& b, const v8f& c) {
  return __builtin_amdgcn_wmma_f32_16x16x32_f16(false, a, false, b, (short)0, c,
                                                false, false);
}

// ---------------------------------------------------------------------------
// PE stage 1: bilinear 1024->64 downsample (align_corners=False, scale 16 ->
// average of the 2x2 input block at offset +7).
// ---------------------------------------------------------------------------
__global__ void k_downsample(const float* __restrict__ x, float* __restrict__ xd) {
  int p = blockIdx.x * blockDim.x + threadIdx.x;  // 0..4095
  int i = p >> 6, j = p & 63;
  const float* r0 = x + (size_t)(i * 16 + 7) * 1024;
  const float* r1 = x + (size_t)(i * 16 + 8) * 1024;
  int c0 = j * 16 + 7, c1 = j * 16 + 8;
  xd[p] = 0.25f * (r0[c0] + r0[c1] + r1[c0] + r1[c1]);
}

// PE stage 2: 3x3 conv (1 -> 256 ch, zero pad) + ReLU, f16 out [256][4096]
__global__ void k_conv3relu(const float* __restrict__ xd,
                            const float* __restrict__ w9,
                            const float* __restrict__ b,
                            _Float16* __restrict__ act) {
  int gid = blockIdx.x * blockDim.x + threadIdx.x;  // 256*4096
  int o = gid >> 12, p = gid & 4095;
  int i = p >> 6, j = p & 63;
  const float* wk = w9 + o * 9;
  float s = b[o];
#pragma unroll
  for (int di = -1; di <= 1; ++di)
#pragma unroll
    for (int dj = -1; dj <= 1; ++dj) {
      int ii = i + di, jj = j + dj;
      if (ii >= 0 && ii < 64 && jj >= 0 && jj < 64)
        s += wk[(di + 1) * 3 + (dj + 1)] * xd[ii * 64 + jj];
    }
  act[(size_t)o * NPIX + p] = (_Float16)fmaxf(s, 0.0f);
}

// ---------------------------------------------------------------------------
// Generic WMMA GEMM: out = epilogue(W[M][K] * X[K][4096] + bias)
// Block = 128 threads (4 waves); tile 64(M) x 128(N); K step 32.
// Wave tile = 16(M) x 128(N) -> 8 accumulators, A reused across 8 WMMAs.
// ---------------------------------------------------------------------------
enum { MODE_F32 = 0, MODE_F16 = 1, MODE_QSCALE = 2, MODE_GELU = 3, MODE_RES = 4 };

__global__ __launch_bounds__(128) void k_gemm_wmma(
    const float* __restrict__ W, const _Float16* __restrict__ X,
    const float* __restrict__ bias, const float* __restrict__ resid,
    const float* __restrict__ scale_ptr,
    float* __restrict__ outF, _Float16* __restrict__ outH,
    int K, int mode) {
  __shared__ _Float16 Wt[64 * ROWH];    // [m][K]  fragment-major
  __shared__ _Float16 Xtt[128 * ROWH];  // [n][K]  fragment-major (transposed)
  int lane = threadIdx.x & 31, w = threadIdx.x >> 5;
  int n0 = blockIdx.x * 128, m0 = blockIdx.y * 64;

  v8f acc[8] = {};
  for (int k0 = 0; k0 < K; k0 += 32) {
    // ---- stage W (f32 -> f16): 64x32 halves, vector loads + vector stores
#pragma unroll
    for (int u = 0; u < 2; ++u) {
      int cid = threadIdx.x + u * 128;        // 256 chunks of 8 halves
      int row = cid >> 2, c0 = (cid & 3) * 8;
      const float* src = W + (size_t)(m0 + row) * K + k0 + c0;
      v4f a0 = *(const v4f*)src;
      v4f a1 = *(const v4f*)(src + 4);
      v8h hh;
#pragma unroll
      for (int j = 0; j < 4; ++j) { hh[j] = (_Float16)a0[j]; hh[4 + j] = (_Float16)a1[j]; }
      *(v8h*)(&Wt[row * ROWH + c0]) = hh;
    }
    // ---- stage X transposed: 32(K) x 128(N) -> Xtt[n][K]
#pragma unroll
    for (int u = 0; u < 4; ++u) {
      int cid = threadIdx.x + u * 128;        // 512 chunks of 8 halves
      int k = cid >> 4, c0 = (cid & 15) * 8;
      v8h xv = *(const v8h*)(X + (size_t)(k0 + k) * NPIX + n0 + c0);
#pragma unroll
      for (int j = 0; j < 8; ++j) Xtt[(c0 + j) * ROWH + k] = xv[j];
    }
    if (k0 + 32 < K)  // global_prefetch_b8 of next X K-slab
      __builtin_prefetch(X + (size_t)(k0 + 32) * NPIX + n0 + threadIdx.x * 8, 0, 3);
    __syncthreads();

    v16h a = load_frag_row(&Wt[(w * 16 + (lane & 15)) * ROWH], lane);
#pragma unroll
    for (int t = 0; t < 8; ++t) {
      v16h b = load_frag_row(&Xtt[(t * 16 + (lane & 15)) * ROWH], lane);
      acc[t] = wmma_f16(a, b, acc[t]);
    }
    __syncthreads();
  }

  float qs = 1.0f;
  if (mode == MODE_QSCALE) qs = 1.0f / scale_ptr[0];
#pragma unroll
  for (int t = 0; t < 8; ++t)
#pragma unroll
    for (int r = 0; r < 8; ++r) {
      int mrow = m0 + w * 16 + r + ((lane & 16) ? 8 : 0);
      int ncol = n0 + t * 16 + (lane & 15);
      size_t o = (size_t)mrow * NPIX + ncol;
      float v = acc[t][r] + bias[mrow];
      if (mode == MODE_QSCALE) v *= qs;
      if (mode == MODE_GELU)   v = 0.5f * v * (1.0f + erff(v * 0.70710678118f));
      if (mode == MODE_RES)    v += resid[o];
      if (outH) outH[o] = (_Float16)v;
      if (outF) outF[o] = v;
    }
}

// ---------------------------------------------------------------------------
// Flash attention: Q,K,V f16 [head][32][4096] (Q pre-scaled by 1/sqrt(hd)).
// Block = 4 waves, each wave owns a 16-query tile; 32-key blocks streamed
// through LDS in fragment-major order (coalesced 16B global reads).
// ---------------------------------------------------------------------------
__global__ __launch_bounds__(128) void k_attn_wmma(
    const _Float16* __restrict__ Q, const _Float16* __restrict__ Kg,
    const _Float16* __restrict__ V, _Float16* __restrict__ O) {
  __shared__ _Float16 Kt[32 * ROWH];      // [key][d]
  __shared__ _Float16 Vt[32 * ROWH];      // [d][key]
  __shared__ _Float16 P[4][16 * ROWH];    // per-wave [query][key]
  int lane = threadIdx.x & 31, w = threadIdx.x >> 5;
  int h = blockIdx.y;
  int q0 = blockIdx.x * 64 + w * 16;
  const _Float16* Qh = Q + (size_t)h * HD * NPIX;
  const _Float16* Kh = Kg + (size_t)h * HD * NPIX;
  const _Float16* Vh = V + (size_t)h * HD * NPIX;

  // A fragment: rows = queries, K = head_dim (exactly 32); loaded once.
  int hi = (lane & 16) ? 8 : 0;
  v16h aq;
#pragma unroll
  for (int e = 0; e < 16; ++e) {
    int d = (e & 7) + hi + ((e & 8) ? 16 : 0);
    aq[e] = Qh[(size_t)d * NPIX + q0 + (lane & 15)];
  }

  float mrow[8], lrow[8];
  v8f o0 = {}, o1 = {};
#pragma unroll
  for (int r = 0; r < 8; ++r) { mrow[r] = -INFINITY; lrow[r] = 0.0f; }

  int sd = threadIdx.x >> 2;            // 0..31: d (or key) row for staging
  int sm = (threadIdx.x & 3) * 8;       // 8-half chunk within 32
  for (int mb = 0; mb < NPIX; mb += 32) {
    // ---- stage K (transposed scatter) and V (vector copy)
    v8h kv = *(const v8h*)(Kh + (size_t)sd * NPIX + mb + sm);
#pragma unroll
    for (int j = 0; j < 8; ++j) Kt[(sm + j) * ROWH + sd] = kv[j];
    v8h vv = *(const v8h*)(Vh + (size_t)sd * NPIX + mb + sm);
    *(v8h*)(&Vt[sd * ROWH + sm]) = vv;
    __syncthreads();

    // ---- scores: S[16q x 32k] as two 16x16 WMMAs over K=hd=32
    v8f z = {};
    v16h bk0 = load_frag_row(&Kt[(lane & 15) * ROWH], lane);
    v8f s0 = wmma_f16(aq, bk0, z);
    v16h bk1 = load_frag_row(&Kt[(16 + (lane & 15)) * ROWH], lane);
    v8f s1 = wmma_f16(aq, bk1, z);

    // ---- online softmax (query row spread over a 16-lane group)
#pragma unroll
    for (int r = 0; r < 8; ++r) {
      float mx = fmaxf(s0[r], s1[r]);
#pragma unroll
      for (int off = 1; off < 16; off <<= 1)
        mx = fmaxf(mx, __shfl_xor(mx, off, 32));
      float nm = fmaxf(mrow[r], mx);
      float f = __expf(mrow[r] - nm);
      mrow[r] = nm;
      lrow[r] *= f;
      o0[r] *= f;
      o1[r] *= f;
      float p0 = __expf(s0[r] - nm), p1 = __expf(s1[r] - nm);
      lrow[r] += p0 + p1;
      int qr = r + ((lane & 16) ? 8 : 0);
      P[w][qr * ROWH + (lane & 15)]      = (_Float16)p0;
      P[w][qr * ROWH + 16 + (lane & 15)] = (_Float16)p1;
    }
    __syncthreads();

    // ---- O += P[16x32] * V^T[32x16d] x2 (d columns 0-15, 16-31)
    v16h ap  = load_frag_row(&P[w][(lane & 15) * ROWH], lane);
    v16h bv0 = load_frag_row(&Vt[(lane & 15) * ROWH], lane);
    o0 = wmma_f16(ap, bv0, o0);
    v16h bv1 = load_frag_row(&Vt[(16 + (lane & 15)) * ROWH], lane);
    o1 = wmma_f16(ap, bv1, o1);
    __syncthreads();
  }

  // ---- finalize: divide by row sums, write fused [head*32+d][n]
#pragma unroll
  for (int r = 0; r < 8; ++r) {
    float ls = lrow[r];
#pragma unroll
    for (int off = 1; off < 16; off <<= 1) ls += __shfl_xor(ls, off, 32);
    float inv = 1.0f / ls;
    int qr = r + ((lane & 16) ? 8 : 0);
    O[(size_t)(h * HD + (lane & 15)) * NPIX + q0 + qr]      = (_Float16)(o0[r] * inv);
    O[(size_t)(h * HD + 16 + (lane & 15)) * NPIX + q0 + qr] = (_Float16)(o1[r] * inv);
  }
}

// ---------------------------------------------------------------------------
// Channel LayerNorms (over 256 channels per pixel).
// ---------------------------------------------------------------------------
__global__ void k_ln_add(const float* __restrict__ rgb, const float* __restrict__ dpt,
                         const float* __restrict__ gr, const float* __restrict__ br,
                         const float* __restrict__ gd, const float* __restrict__ bd,
                         const float* __restrict__ pos, const float* __restrict__ pe,
                         _Float16* __restrict__ rgbH, _Float16* __restrict__ dptH) {
  int p = blockIdx.x * blockDim.x + threadIdx.x;
  float s1 = 0, s2 = 0, t1 = 0, t2 = 0;
  for (int c = 0; c < DIMC; ++c) {
    float a = rgb[(size_t)c * NPIX + p]; s1 += a; s2 += a * a;
    float b = dpt[(size_t)c * NPIX + p]; t1 += b; t2 += b * b;
  }
  float mu1 = s1 / DIMC, r1 = rsqrtf(s2 / DIMC - mu1 * mu1 + LN_EPS);
  float mu2 = t1 / DIMC, r2 = rsqrtf(t2 / DIMC - mu2 * mu2 + LN_EPS);
  for (int c = 0; c < DIMC; ++c) {
    size_t o = (size_t)c * NPIX + p;
    float add = pos[c] + pe[o];
    rgbH[o] = (_Float16)((rgb[o] - mu1) * r1 * gr[c] + br[c] + add);
    dptH[o] = (_Float16)((dpt[o] - mu2) * r2 * gd[c] + bd[c] + add);
  }
}

__global__ void k_ln(const float* __restrict__ x,
                     const float* __restrict__ g, const float* __restrict__ b,
                     _Float16* __restrict__ outH, float* __restrict__ outF) {
  int p = blockIdx.x * blockDim.x + threadIdx.x;
  float s1 = 0, s2 = 0;
  for (int c = 0; c < DIMC; ++c) {
    float a = x[(size_t)c * NPIX + p]; s1 += a; s2 += a * a;
  }
  float mu = s1 / DIMC, rs = rsqrtf(s2 / DIMC - mu * mu + LN_EPS);
  for (int c = 0; c < DIMC; ++c) {
    size_t o = (size_t)c * NPIX + p;
    float v = (x[o] - mu) * rs * g[c] + b[c];
    if (outH) outH[o] = (_Float16)v;
    if (outF) outF[o] = v;
  }
}

// ---------------------------------------------------------------------------
// Launch
// ---------------------------------------------------------------------------
extern "C" void kernel_launch(void* const* d_in, const int* in_sizes, int n_in,
                              void* d_out, int out_size, void* d_ws, size_t ws_size,
                              hipStream_t stream) {
  (void)in_sizes; (void)n_in; (void)out_size; (void)ws_size;
  const float* rgb_feat = (const float*)d_in[0];
  const float* dpt_feat = (const float*)d_in[1];
  const float* x_dpt    = (const float*)d_in[2];
  const float* wq = (const float*)d_in[3];  const float* bq = (const float*)d_in[4];
  const float* wk = (const float*)d_in[5];  const float* bk = (const float*)d_in[6];
  const float* wv = (const float*)d_in[7];  const float* bv = (const float*)d_in[8];
  const float* wo = (const float*)d_in[9];  const float* bo = (const float*)d_in[10];
  const float* g_rgb = (const float*)d_in[11]; const float* b_rgb = (const float*)d_in[12];
  const float* g_dpt = (const float*)d_in[13]; const float* b_dpt = (const float*)d_in[14];
  const float* g1 = (const float*)d_in[15]; const float* b1 = (const float*)d_in[16];
  const float* g2 = (const float*)d_in[17]; const float* b2 = (const float*)d_in[18];
  const float* w_mlp1 = (const float*)d_in[19]; const float* b_mlp1 = (const float*)d_in[20];
  const float* w_mlp2 = (const float*)d_in[21]; const float* b_mlp2 = (const float*)d_in[22];
  const float* w_de1 = (const float*)d_in[23]; const float* b_de1 = (const float*)d_in[24];
  const float* w_de2 = (const float*)d_in[25]; const float* b_de2 = (const float*)d_in[26];
  const float* pos_emb = (const float*)d_in[27];
  const float* scale = (const float*)d_in[28];
  float* out = (float*)d_out;

  char* ws = (char*)d_ws;
  size_t off = 0;
  auto alloc = [&](size_t bytes) {
    size_t o = off; off += (bytes + 255) & ~(size_t)255; return o;
  };
  float*    xd      = (float*)   (ws + alloc(NPIX * 4));
  _Float16* act1    = (_Float16*)(ws + alloc((size_t)DIMC * NPIX * 2));
  float*    pe      = (float*)   (ws + alloc((size_t)DIMC * NPIX * 4));
  _Float16* rgbln   = (_Float16*)(ws + alloc((size_t)DIMC * NPIX * 2));
  _Float16* dptln   = (_Float16*)(ws + alloc((size_t)DIMC * NPIX * 2));
  _Float16* qh      = (_Float16*)(ws + alloc((size_t)DIMC * NPIX * 2));
  _Float16* kh      = (_Float16*)(ws + alloc((size_t)DIMC * NPIX * 2));
  _Float16* vh      = (_Float16*)(ws + alloc((size_t)DIMC * NPIX * 2));
  _Float16* fused   = (_Float16*)(ws + alloc((size_t)DIMC * NPIX * 2));
  float*    outf    = (float*)   (ws + alloc((size_t)DIMC * NPIX * 4));
  _Float16* outlnh  = (_Float16*)(ws + alloc((size_t)DIMC * NPIX * 2));
  float*    outlnf  = (float*)   (ws + alloc((size_t)DIMC * NPIX * 4));
  _Float16* hh      = (_Float16*)(ws + alloc((size_t)MLPC * NPIX * 2));
  float*    out2    = (float*)   (ws + alloc((size_t)DIMC * NPIX * 4));

  dim3 blk128(128), blk256(256);
  dim3 gemm256(NPIX / 128, DIMC / 64);   // M=256
  dim3 gemm1024(NPIX / 128, MLPC / 64);  // M=1024

  // 1. depth positional embedding
  k_downsample<<<NPIX / 256, blk256, 0, stream>>>(x_dpt, xd);
  k_conv3relu<<<(DIMC * NPIX) / 256, blk256, 0, stream>>>(xd, w_de1, b_de1, act1);
  k_gemm_wmma<<<gemm256, blk128, 0, stream>>>(w_de2, act1, b_de2, nullptr, nullptr,
                                              pe, nullptr, DIMC, MODE_F32);
  // 2. LayerNorms + pos_emb + pe -> f16 activations
  k_ln_add<<<NPIX / 256, blk256, 0, stream>>>(rgb_feat, dpt_feat, g_rgb, b_rgb,
                                              g_dpt, b_dpt, pos_emb, pe, rgbln, dptln);
  // 3. Q (scaled by 1/sqrt(hd)), K, V projections
  k_gemm_wmma<<<gemm256, blk128, 0, stream>>>(wq, rgbln, bq, nullptr, scale,
                                              nullptr, qh, DIMC, MODE_QSCALE);
  k_gemm_wmma<<<gemm256, blk128, 0, stream>>>(wk, dptln, bk, nullptr, nullptr,
                                              nullptr, kh, DIMC, MODE_F16);
  k_gemm_wmma<<<gemm256, blk128, 0, stream>>>(wv, dptln, bv, nullptr, nullptr,
                                              nullptr, vh, DIMC, MODE_F16);
  // 4. flash attention
  k_attn_wmma<<<dim3(NPIX / 64, HEADS), blk128, 0, stream>>>(qh, kh, vh, fused);
  // 5. output projection + residual
  k_gemm_wmma<<<gemm256, blk128, 0, stream>>>(wo, fused, bo, rgb_feat, nullptr,
                                              outf, nullptr, DIMC, MODE_RES);
  // 6. LN1
  k_ln<<<NPIX / 256, blk256, 0, stream>>>(outf, g1, b1, outlnh, outlnf);
  // 7. MLP
  k_gemm_wmma<<<gemm1024, blk128, 0, stream>>>(w_mlp1, outlnh, b_mlp1, nullptr, nullptr,
                                               nullptr, hh, DIMC, MODE_GELU);
  k_gemm_wmma<<<gemm256, blk128, 0, stream>>>(w_mlp2, hh, b_mlp2, outlnf, nullptr,
                                              out2, nullptr, MLPC, MODE_RES);
  // 8. final LN -> d_out
  k_ln<<<NPIX / 256, blk256, 0, stream>>>(out2, g2, b2, nullptr, out);
}
